// LatticeOutput_69870527971628
// MI455X (gfx1250) — matrix operand order repaired and checked by reading, not compiled, not run on hardware
//
#include <hip/hip_runtime.h>
#include <hip/hip_bf16.h>

typedef __attribute__((ext_vector_type(16))) _Float16 v16h;
typedef __attribute__((ext_vector_type(8)))  _Float16 v8h;
typedef __attribute__((ext_vector_type(8)))  float    v8f;
typedef __attribute__((ext_vector_type(4)))  float    v4f;

#define C_DIM 256
#define B_SEG 4096
#define OUTC  6
#define ROWS_PER_WAVE 256

// ---------------------------------------------------------------------------
// Kernel 1: zero the accumulators (seg_sum [B,C] followed contiguously by counts [B])
// ---------------------------------------------------------------------------
__global__ void zero_ws_kernel(float* __restrict__ p, int n) {
    int i = blockIdx.x * blockDim.x + threadIdx.x;
    int stride = gridDim.x * blockDim.x;
    for (; i < n; i += stride) p[i] = 0.0f;
}

// ---------------------------------------------------------------------------
// Kernel 2: segment sum exploiting sorted batch_vec.
// One wave32 owns a 256-row chunk; lane t owns columns [8t, 8t+8) so one wave
// covers a full 256-float row with two b128 loads per lane (1 KB / request
// pair -> 4x fewer VMEM ops than b32-per-lane on the 512 MB stream).
// Registers accumulate while the (wave-uniform, scalar-loaded) segment id is
// unchanged; each run flushes with 8 float atomics per lane. Non-temporal
// loads (stream > L2) + global_prefetch_b8 running 16 rows ahead.
// ---------------------------------------------------------------------------
__global__ void __launch_bounds__(256) seg_sum_kernel(
    const float* __restrict__ s, const int* __restrict__ bv,
    float* __restrict__ sums, float* __restrict__ cnts, int M) {
    const int lane = threadIdx.x & 31;
    const int wid  = threadIdx.x >> 5;
    const int wg   = blockIdx.x * 8 + wid;        // global wave id
    int r0 = wg * ROWS_PER_WAVE;
    if (r0 >= M) return;
    int r1 = min(M, r0 + ROWS_PER_WAVE);
    const int c0 = lane * 8;

    int   cur    = bv[r0];                        // wave-uniform -> scalar load
    int   runlen = 0;
    v4f acc0 = {0.f, 0.f, 0.f, 0.f};
    v4f acc1 = {0.f, 0.f, 0.f, 0.f};

    for (int r = r0; r < r1; ++r) {
        int seg = bv[r];                          // uniform
        if (seg != cur) {
            float* dst = sums + (size_t)cur * C_DIM + c0;
#pragma unroll
            for (int j = 0; j < 4; ++j) {
                atomicAdd(dst + j,     acc0[j]);
                atomicAdd(dst + 4 + j, acc1[j]);
            }
            if (lane == 0) atomicAdd(&cnts[cur], (float)runlen);
            acc0 = (v4f){0.f, 0.f, 0.f, 0.f};
            acc1 = (v4f){0.f, 0.f, 0.f, 0.f};
            runlen = 0;
            cur = seg;
        }
        const v4f* row = (const v4f*)(s + (size_t)r * C_DIM + c0);
        __builtin_prefetch(s + (size_t)(r + 16) * C_DIM + c0, 0, 0);  // global_prefetch_b8
        acc0 += __builtin_nontemporal_load(row);
        acc1 += __builtin_nontemporal_load(row + 1);
        ++runlen;
    }
    float* dst = sums + (size_t)cur * C_DIM + c0;
#pragma unroll
    for (int j = 0; j < 4; ++j) {
        atomicAdd(dst + j,     acc0[j]);
        atomicAdd(dst + 4 + j, acc1[j]);
    }
    if (lane == 0) atomicAdd(&cnts[cur], (float)runlen);
}

// ---------------------------------------------------------------------------
// Kernel 3: mean (f32 -> f16), W1 -> f16, W2 -> f16 padded to 16 rows.
// ---------------------------------------------------------------------------
__global__ void pack_kernel(const float* __restrict__ sums, const float* __restrict__ cnts,
                            const float* __restrict__ W1, const float* __restrict__ W2,
                            _Float16* __restrict__ meanh, _Float16* __restrict__ w1h,
                            _Float16* __restrict__ w2h) {
    const int NM  = B_SEG * C_DIM;   // 1048576
    const int NW1 = C_DIM * C_DIM;   // 65536
    const int NW2 = 16 * C_DIM;      // 4096 (padded)
    int i = blockIdx.x * blockDim.x + threadIdx.x;
    if (i < NM) {
        float cnt = cnts[i >> 8];
        float inv = 1.0f / fmaxf(cnt, 1.0f);
        meanh[i] = (_Float16)(sums[i] * inv);
    } else if (i < NM + NW1) {
        int j = i - NM;
        w1h[j] = (_Float16)W1[j];
    } else if (i < NM + NW1 + NW2) {
        int j = i - NM - NW1;
        int n = j >> 8;
        w2h[j] = (_Float16)((n < OUTC) ? W2[j] : 0.0f);
    }
}

// ---------------------------------------------------------------------------
// Kernel 4: WMMA MLP. One wave per 16-row tile of the [4096,256] mean matrix.
// GEMM1: h = relu(mean @ W1^T + b1), 16x16 output tiles via v_wmma_f32_16x16x32_f16.
// h goes through LDS (C/D layout -> A layout transpose). GEMM2 against padded
// W2^T, softplus epilogue on the 6 valid output columns.
// ---------------------------------------------------------------------------
__device__ __forceinline__ v16h cat8(v8h lo, v8h hi) {
    return __builtin_shufflevector(lo, hi, 0,1,2,3,4,5,6,7,8,9,10,11,12,13,14,15);
}

__global__ void __launch_bounds__(128) mlp_wmma_kernel(
    const _Float16* __restrict__ meanh, const _Float16* __restrict__ w1h,
    const _Float16* __restrict__ w2h, const float* __restrict__ b1,
    const float* __restrict__ b2, float* __restrict__ out) {
    __shared__ _Float16 hbuf[4 * 16 * C_DIM];   // 32 KB: one 16x256 f16 tile per wave

    const int lane  = threadIdx.x & 31;
    const int wid   = threadIdx.x >> 5;
    const int half  = lane >> 4;     // K-half selector
    const int idx16 = lane & 15;     // A: row m; B: col n; D: col n
    const int tile  = blockIdx.x * 4 + wid;
    const int row0  = tile * 16;

    _Float16* hl = hbuf + wid * 16 * C_DIM;

    // ---- GEMM1: [16,256] x [256,256] -> h [16,256], ReLU, stash in LDS ----
    const v8h* Arow = (const v8h*)(meanh + (size_t)(row0 + idx16) * C_DIM);
    for (int n0 = 0; n0 < C_DIM; n0 += 16) {
        const v8h* Brow = (const v8h*)(w1h + (size_t)(n0 + idx16) * C_DIM);
        v8f acc = {};
#pragma unroll
        for (int k0 = 0; k0 < C_DIM; k0 += 32) {
            // A frag: 16 contiguous halves at k0 + 16*half
            int ka = (k0 >> 3) + 2 * half;
            v16h a = cat8(Arow[ka], Arow[ka + 1]);
            // B frag: 16 contiguous halves at k0 + 16*half
            int kb = (k0 >> 3) + 2 * half;
            v16h b = cat8(Brow[kb], Brow[kb + 1]);
            acc = __builtin_amdgcn_wmma_f32_16x16x32_f16(
                false, a, false, b, (short)0, acc, false, false);
        }
        float bias = b1[n0 + idx16];
#pragma unroll
        for (int v = 0; v < 8; ++v) {
            int m = v + 8 * half;                 // C/D layout row
            float hv = fmaxf(acc[v] + bias, 0.0f);
            hl[m * C_DIM + n0 + idx16] = (_Float16)hv;
        }
    }

    // ---- GEMM2: h [16,256] x W2^T [256,16(pad)] -> [16,6], softplus ----
    const v8h* Hrow  = (const v8h*)(hl + (size_t)idx16 * C_DIM);
    const v8h* B2row = (const v8h*)(w2h + (size_t)idx16 * C_DIM);
    v8f acc2 = {};
#pragma unroll
    for (int k0 = 0; k0 < C_DIM; k0 += 32) {
        int ka = (k0 >> 3) + 2 * half;
        v16h a = cat8(Hrow[ka], Hrow[ka + 1]);
        int kb = (k0 >> 3) + 2 * half;
        v16h b = cat8(B2row[kb], B2row[kb + 1]);
        acc2 = __builtin_amdgcn_wmma_f32_16x16x32_f16(
            false, a, false, b, (short)0, acc2, false, false);
    }
    if (idx16 < OUTC) {
        float bias = b2[idx16];
#pragma unroll
        for (int v = 0; v < 8; ++v) {
            int m = v + 8 * half;
            float x  = acc2[v] + bias;
            // numerically stable softplus: max(x,0) + log1p(exp(-|x|))
            float sp = fmaxf(x, 0.0f) + __logf(1.0f + __expf(-fabsf(x)));
            out[(size_t)(row0 + m) * OUTC + idx16] = sp;
        }
    }
}

// ---------------------------------------------------------------------------
extern "C" void kernel_launch(void* const* d_in, const int* in_sizes, int n_in,
                              void* d_out, int out_size, void* d_ws, size_t ws_size,
                              hipStream_t stream) {
    (void)n_in; (void)out_size; (void)ws_size;
    const float* s  = (const float*)d_in[0];
    const int*   bv = (const int*)d_in[1];
    const float* W1 = (const float*)d_in[2];
    const float* b1 = (const float*)d_in[3];
    const float* W2 = (const float*)d_in[4];
    const float* b2 = (const float*)d_in[5];
    float* out = (float*)d_out;
    const int M = in_sizes[1];   // batch_vec length

    // workspace layout (all 16B aligned):
    //   sums  : float   [B_SEG*C_DIM]          (4 MB)
    //   cnts  : float   [B_SEG]                (16 KB, contiguous after sums)
    //   meanh : f16     [B_SEG*C_DIM]          (2 MB)
    //   w1h   : f16     [C_DIM*C_DIM]          (128 KB)
    //   w2h   : f16     [16*C_DIM]             (8 KB, zero-padded rows 6..15)
    char* ws = (char*)d_ws;
    float*    sums  = (float*)ws;
    float*    cnts  = sums + (size_t)B_SEG * C_DIM;
    _Float16* meanh = (_Float16*)(ws + (size_t)B_SEG * C_DIM * 4 + (size_t)B_SEG * 4);
    _Float16* w1h   = meanh + (size_t)B_SEG * C_DIM;
    _Float16* w2h   = w1h + C_DIM * C_DIM;

    zero_ws_kernel<<<512, 256, 0, stream>>>(sums, B_SEG * C_DIM + B_SEG);

    int nwaves = (M + ROWS_PER_WAVE - 1) / ROWS_PER_WAVE;   // 1954
    int nblk   = (nwaves + 7) / 8;                          // 8 waves / block
    seg_sum_kernel<<<nblk, 256, 0, stream>>>(s, bv, sums, cnts, M);

    int total = B_SEG * C_DIM + C_DIM * C_DIM + 16 * C_DIM;
    pack_kernel<<<(total + 255) / 256, 256, 0, stream>>>(sums, cnts, W1, W2, meanh, w1h, w2h);

    mlp_wmma_kernel<<<B_SEG / 16 / 4, 128, 0, stream>>>(meanh, w1h, w2h, b1, b2, out);
}
